// OptimisationLayer_4741643895656
// MI455X (gfx1250) — matrix operand order repaired and checked by reading, not compiled, not run on hardware
//
#include <hip/hip_runtime.h>

// ---------------------------------------------------------------------------
// PDHG LP solver for MI455X (gfx1250): persistent 16-sample tiles in LDS,
// both per-iteration matmuls done with V_WMMA_F32_16X16X32_F16.
// Tile counts are padded so every wave owns a fixed tile set -> zero divergent
// control flow around WMMA. K-fragment loads use a per-iteration k-rotation so
// LICM cannot hoist them (they stay global_load_b128 hitting WGP$/L2).
// ---------------------------------------------------------------------------

typedef __attribute__((ext_vector_type(16))) _Float16 v16h;
typedef __attribute__((ext_vector_type(8)))  _Float16 v8h;
typedef __attribute__((ext_vector_type(8)))  float    v8f;

#define T48   48
#define NVAR  289      // 6*T+1 primal vars
#define MCON  625      // (T+1) + 12*T dual vars
#define NXP   384      // primal width padded to 24 tiles (3 per wave, 8 waves)
#define NXT   24       // NXP/16
#define KS2   10       // matmul2 k-steps: 289 padded to 320 = 10*32
#define NY    640      // dual width padded (40 tiles, 20 k-steps)
#define NYT   40       // NY/16
#define KS1   20       // NY/32
#define NEQ   49       // T+1 (no clamp on first NEQ duals)

#define F1_ELEMS (NXT * KS1 * 512)   // matmul1 B-fragments (K)
#define F2_ELEMS (NYT * KS2 * 512)   // matmul2 B-fragments (K^T)

// ---- K(r,c) computed analytically from the reference construction ---------
__device__ __forceinline__ float Kval(int r, int c) {
    const int OI = 49, OO = 97, OPL = 145, OPR = 193, OE = 241;
    if (r < T48) {
        if (c == r + 1)  return  1.0f;
        if (c == r)      return -0.999f;   // -A_SC
        if (c == OI + r) return -0.95f;    // -B00
        if (c == OO + r) return  1.05f;    // -B01
        return 0.0f;
    }
    if (r == T48) return (c == 0) ? 1.0f : 0.0f;
    int g = r - T48 - 1, b = g / T48, t = g % T48;
    float s = 0.0f;
    switch (b) {
        case 0:  if (c == t)        s = -1.0f; break;
        case 1:  if (c == t)        s =  1.0f; break;
        case 2:  if (c == OI + t)   s =  1.0f; break;
        case 3:  if (c == OI + t)   s = -1.0f; break;
        case 4:  if (c == OO + t)   s =  1.0f; break;
        case 5:  if (c == OO + t)   s = -1.0f; break;
        case 6:  if (c == OPL + t)  s = -1.0f;
                 if (c == OE  + t)  s = -1.0f; break;
        case 7:  if (c == OPL + t)  s =  1.0f;
                 if (c == OE  + t)  s = -1.0f; break;
        case 8:  if (c == OPR + t)  s = -1.0f;
                 if (c == OE  + t)  s = -1.0f; break;
        case 9:  if (c == OPR + t)  s =  1.0f;
                 if (c == OE  + t)  s = -1.0f; break;
        case 10: if (c == OPR + t)  s = -1.0f;
                 if (c == OPL + t)  s =  1.0f;
                 if (c == OI  + t)  s =  1.0f;
                 if (c == OO  + t)  s = -1.0f; break;
        case 11: if (c == OPR + t)  s =  1.0f;
                 if (c == OPL + t)  s = -1.0f;
                 if (c == OI  + t)  s = -1.0f;
                 if (c == OO  + t)  s =  1.0f; break;
    }
    return s;
}

// ---- ||K||_2 by power iteration (device-side, graph-capture safe) ---------
__global__ __launch_bounds__(256) void knorm_kernel(float* out_knorm) {
    __shared__ float v[NVAR];
    __shared__ float w[MCON];
    __shared__ float red[256];
    int tid = threadIdx.x;
    for (int c = tid; c < NVAR; c += 256) v[c] = 1.0f;
    __syncthreads();
    float norm2 = 1.0f;
    for (int it = 0; it < 64; ++it) {
        for (int r = tid; r < MCON; r += 256) {
            float s = 0.0f;
            for (int c = 0; c < NVAR; ++c) s += Kval(r, c) * v[c];
            w[r] = s;
        }
        __syncthreads();
        for (int c = tid; c < NVAR; c += 256) {
            float s = 0.0f;
            for (int r = 0; r < MCON; ++r) s += Kval(r, c) * w[r];
            v[c] = s;
        }
        __syncthreads();
        float loc = 0.0f;
        for (int c = tid; c < NVAR; c += 256) loc += v[c] * v[c];
        red[tid] = loc;
        __syncthreads();
        for (int s = 128; s > 0; s >>= 1) {
            if (tid < s) red[tid] += red[tid + s];
            __syncthreads();
        }
        norm2 = red[0];
        float inv = rsqrtf(norm2);
        __syncthreads();
        for (int c = tid; c < NVAR; c += 256) v[c] *= inv;
        __syncthreads();
    }
    // ||K^T K v|| -> sigma_max^2 ; norm2 is its square.
    if (tid == 0) out_knorm[0] = sqrtf(sqrtf(norm2));
}

// ---- Build pre-swizzled f16 B-fragments for WMMA (32x16 per-lane layout) --
// B frag element index e = lane*16 + j ; h=lane/16: j<8 -> k=8h+j, j>=8 -> k=16+8h+(j-8),
// column n = lane%16  (per CDNA5 ISA 16-bit B layout).
__global__ __launch_bounds__(256) void build_frags(_Float16* B1, _Float16* B2) {
    int total = F1_ELEMS + F2_ELEMS;
    for (int i = blockIdx.x * blockDim.x + threadIdx.x; i < total;
         i += gridDim.x * blockDim.x) {
        if (i < F1_ELEMS) {
            int f = i, tile = f >> 9, e = f & 511;
            int ks = tile % KS1, nt = tile / KS1;
            int lane = e >> 4, j = e & 15;
            int kk = (j < 8) ? ((lane >> 4) << 3) + j : 16 + ((lane >> 4) << 3) + (j - 8);
            int nn = lane & 15;
            int k = ks * 32 + kk;          // row of K (dual dim, 625)
            int n = nt * 16 + nn;          // col of K (primal dim, 289)
            float val = (k < MCON && n < NVAR) ? Kval(k, n) : 0.0f;
            B1[f] = (_Float16)val;
        } else {
            int f = i - F1_ELEMS, tile = f >> 9, e = f & 511;
            int ks = tile % KS2, nt = tile / KS2;
            int lane = e >> 4, j = e & 15;
            int kk = (j < 8) ? ((lane >> 4) << 3) + j : 16 + ((lane >> 4) << 3) + (j - 8);
            int nn = lane & 15;
            int k = ks * 32 + kk;          // row of K^T (primal dim, 289)
            int n = nt * 16 + nn;          // col of K^T (dual dim, 625)
            float val = (k < NVAR && n < MCON) ? Kval(n, k) : 0.0f;
            B2[f] = (_Float16)val;
        }
    }
}

// ---- A-fragment load from LDS (16-bit A 16x32 layout: lane = M) -----------
__device__ __forceinline__ v16h load_afrag(const _Float16* row, int kbase, int koff) {
    union { v16h v; v8h h[2]; } u;
    u.h[0] = *(const v8h*)(row + kbase + koff);        // K = kbase+koff .. +7
    u.h[1] = *(const v8h*)(row + kbase + koff + 16);   // K = kbase+koff+16 ..
    return u.v;
}

// ---- Main persistent PDHG kernel: one WG = 16 samples, 300 iterations -----
__global__ __launch_bounds__(256, 1) void pdhg_kernel(
    const float* __restrict__ price, const float* __restrict__ load,
    const float* __restrict__ pv,    const float* __restrict__ s0,
    const float* __restrict__ knorm_ws,
    const _Float16* __restrict__ B1, const _Float16* __restrict__ B2,
    float* __restrict__ out, int Btot) {

    __shared__ __attribute__((aligned(32))) float    xf[16 * NXP];
    __shared__ __attribute__((aligned(32))) float    cf[16 * NXP];
    __shared__ __attribute__((aligned(32))) float    yf[16 * NY];
    __shared__ __attribute__((aligned(32))) float    qf[16 * NY];
    __shared__ __attribute__((aligned(32))) _Float16 yh[16 * NY];
    __shared__ __attribute__((aligned(32))) _Float16 xbh[16 * NXP];

    const int tid = threadIdx.x;
    const int wg  = blockIdx.x;

    // ---- init state + per-sample constant vectors c, q --------------------
    for (int i = tid; i < 16 * NXP; i += 256) {
        int m = i / NXP, col = i % NXP;
        int s = wg * 16 + m;
        xf[i] = 0.0f;
        xbh[i] = (_Float16)0.0f;
        float cv = 0.0f;
        if      (col >= 49  && col < 97)  cv =  price[s * T48 + (col - 49)];
        else if (col >= 97  && col < 145) cv = -price[s * T48 + (col - 97)];
        else if (col >= 145 && col < 193) cv =  price[s * T48 + (col - 145)];
        else if (col >= 193 && col < 241) cv = -price[s * T48 + (col - 193)];
        else if (col >= 241 && col < 289) cv = 10000.0f;   // EPS_W
        cf[i] = cv;
    }
    for (int i = tid; i < 16 * NY; i += 256) {
        int m = i / NY, col = i % NY;
        int s = wg * 16 + m;
        yf[i] = 0.0f;
        yh[i] = (_Float16)0.0f;
        float qv = 0.0f;
        if (col == 48) qv = s0[s];
        else if (col >= 49 && col < 625) {
            int b = (col - 49) / T48, t = (col - 49) % T48;
            switch (b) {
                case 1:  qv = 13.5f;               break;  // SMAX
                case 2:  qv = 5.0f;                break;  // PS_MAX
                case 4:  qv = 5.0f;                break;  // PS_MAX
                case 6:  qv = -load[s * T48 + t];  break;
                case 7:  qv =  load[s * T48 + t];  break;
                case 8:  qv = -pv[s * T48 + t];    break;
                case 9:  qv =  pv[s * T48 + t];    break;
                case 10: qv = 10.0f;               break;  // -PG_MIN
                case 11: qv = 10.0f;               break;  //  PG_MAX
                default: qv = 0.0f;                break;
            }
        }
        qf[i] = qv;
    }
    __syncthreads();

    const float knorm = knorm_ws[0];
    const float tau   = 0.95f / knorm;
    const float sigma = tau;
    const float invd  = 1.0f / (1.0f + tau * 1e-6f);   // 1/(1+tau*DELTA)

    const int lane = tid & 31;
    const int wv   = __builtin_amdgcn_readfirstlane(tid >> 5);  // wave id, SGPR
    const int m_a  = lane & 15;           // A-fragment: M = lane%16
    const int koff = (lane >> 4) << 3;    // A-fragment k split (lanes 16..31)
    const int m_d  = (lane >> 4) << 3;    // D-fragment: M = r + 8*(lane>=16)
    const int n_d  = lane & 15;           // D-fragment: N = lane%16

    // ---- one-time warm-up: pull this wave's K fragments toward the WGP ----
    for (int ks = 0; ks < KS1; ++ks)
        for (int i = 0; i < 3; ++i)
            __builtin_prefetch(B1 + ((((wv + 8 * i) * KS1 + ks) << 9) + (lane << 4)), 0, 1);
    for (int ks = 0; ks < KS2; ++ks)
        for (int i = 0; i < 5; ++i)
            __builtin_prefetch(B2 + ((((wv + 8 * i) * KS2 + ks) << 9) + (lane << 4)), 0, 1);

    int rot1 = 0, rot2 = 0;   // per-iteration k-step rotation (defeats LICM)

    for (int it = 0; it < 300; ++it) {
        // ---------- matmul1: U = y @ K   (16 x 640) . (640 x 384) ----------
        v8f acc[3];
        acc[0] = {}; acc[1] = {}; acc[2] = {};
        const _Float16* yrow = yh + m_a * NY;
#pragma unroll 2
        for (int ks = 0; ks < KS1; ++ks) {
            int ksr = ks + rot1; if (ksr >= KS1) ksr -= KS1;
            v16h a = load_afrag(yrow, ksr * 32, koff);
#pragma unroll
            for (int i = 0; i < 3; ++i) {
                int nt = wv + 8 * i;                       // 24 tiles, 3/wave
                v16h b = *(const v16h*)(B1 + (((nt * KS1 + ksr) << 9) + (lane << 4)));
                acc[i] = __builtin_amdgcn_wmma_f32_16x16x32_f16(
                    false, a, false, b, (short)0, acc[i], false, false);
            }
        }
        // primal update: xn = (x - tau*(c + U)) / (1 + tau*DELTA)
#pragma unroll
        for (int i = 0; i < 3; ++i) {
            int col = (wv + 8 * i) * 16 + n_d;
#pragma unroll
            for (int r = 0; r < 8; ++r) {
                int m = m_d + r;
                int ix = m * NXP + col;
                float xo = xf[ix];
                float xn = (xo - tau * (cf[ix] + acc[i][r])) * invd;
                xf[ix] = xn;
                xbh[ix] = (_Float16)(2.0f * xn - xo);
            }
        }
        __syncthreads();

        // ---------- matmul2: V = (2xn - x) @ K^T  (16 x 320) . (320 x 640) -
        v8f bacc[5];
        bacc[0] = {}; bacc[1] = {}; bacc[2] = {}; bacc[3] = {}; bacc[4] = {};
        const _Float16* xrow = xbh + m_a * NXP;
#pragma unroll 2
        for (int ks = 0; ks < KS2; ++ks) {
            int ksr = ks + rot2; if (ksr >= KS2) ksr -= KS2;
            v16h a = load_afrag(xrow, ksr * 32, koff);
#pragma unroll
            for (int i = 0; i < 5; ++i) {
                int nt = wv + 8 * i;                       // 40 tiles, 5/wave
                v16h b = *(const v16h*)(B2 + (((nt * KS2 + ksr) << 9) + (lane << 4)));
                bacc[i] = __builtin_amdgcn_wmma_f32_16x16x32_f16(
                    false, a, false, b, (short)0, bacc[i], false, false);
            }
        }
        // dual update: yb = y + sigma*V - sigma*q ; clamp tail to >= 0
#pragma unroll
        for (int i = 0; i < 5; ++i) {
            int col = (wv + 8 * i) * 16 + n_d;
#pragma unroll
            for (int r = 0; r < 8; ++r) {
                int m = m_d + r;
                int iy = m * NY + col;
                float yn = yf[iy] + sigma * bacc[i][r] - sigma * qf[iy];
                if (col >= NEQ) yn = fmaxf(yn, 0.0f);
                yf[iy] = yn;
                yh[iy] = (_Float16)yn;
            }
        }
        __syncthreads();

        if (++rot1 == KS1) rot1 = 0;
        if (++rot2 == KS2) rot2 = 0;
    }

    // ---- emit (Ps_in, Ps_out, Pl, Pr, eps) = x[:, 49 : 49+5*48] -----------
    for (int i = tid; i < 5 * 16 * T48; i += 256) {
        int o   = i / (16 * T48);
        int rem = i % (16 * T48);
        int m = rem / T48, t = rem % T48;
        out[(size_t)o * Btot * T48 + (size_t)(wg * 16 + m) * T48 + t] =
            xf[m * NXP + (49 + o * T48 + t)];
    }
}

// ---------------------------------------------------------------------------
extern "C" void kernel_launch(void* const* d_in, const int* in_sizes, int n_in,
                              void* d_out, int out_size, void* d_ws, size_t ws_size,
                              hipStream_t stream) {
    const float* price = (const float*)d_in[0];
    const float* loadp = (const float*)d_in[1];
    const float* pvp   = (const float*)d_in[2];
    const float* s0p   = (const float*)d_in[3];
    float* out = (float*)d_out;

    int Btot = in_sizes[0] / T48;   // B = 1024

    char* ws = (char*)d_ws;
    float*     knorm = (float*)ws;
    _Float16*  B1    = (_Float16*)(ws + 256);
    _Float16*  B2    = B1 + F1_ELEMS;

    knorm_kernel<<<1, 256, 0, stream>>>(knorm);

    int total  = F1_ELEMS + F2_ELEMS;
    int blocks = (total + 255) / 256;
    build_frags<<<blocks, 256, 0, stream>>>(B1, B2);

    pdhg_kernel<<<Btot / 16, 256, 0, stream>>>(price, loadp, pvp, s0p,
                                               knorm, B1, B2, out, Btot);
}
